// EquiGroupSamplingC8_69037304315943
// MI455X (gfx1250) — matrix operand order repaired and checked by reading, not compiled, not run on hardware
//
#include <hip/hip_runtime.h>

// ============================================================================
// EquiGroupSamplingC8 for MI455X (gfx1250, wave32, WMMA, TDM)
//
// Pipeline (all on `stream`, deterministic, graph-capturable):
//   S1  k_build_rot   : fp64 expm of SO(3) generators -> Euler-factor matrices
//   S2  k_build_c8w   : C8W (8 x 455) fp32
//   S3  k_build_grids : D_IN (2048 x 10) fp32, BT3 = D_OUTW^T padded (512 x 2048) f16
//   S4  k_build_T     : fused (proj_2 x C8W) weights TT (512 x 8192) f16
//   S5  k_sig         : proj_1 + forward grid transform + relu*sqrt2 -> sig f16
//   S6  k_gemm_nt     : GEMM1 feat2 = sig(8192x2048)*BT3^T  [WMMA, TDM dbl-buffered,
//                       f16 epilogue]
//   S7  k_gemm_nt     : GEMM2 traj_c8 = feat2(512x8192)*TT^T [WMMA, TDM, split-K 8]
//   S8  k_reduce      : deterministic split-K reduction -> d_out (traj part)
//   S9  k_xc8         : x(65536x455)*C8W^T -> d_out (x part), HBM-bound, prefetch
//
// Workspace requirement: ~63 MB.
// ============================================================================

typedef _Float16 f16;
typedef __attribute__((ext_vector_type(16))) _Float16 v16h;
typedef __attribute__((ext_vector_type(8)))  _Float16 v8h;
typedef __attribute__((ext_vector_type(8)))  float    v8f;
typedef __attribute__((ext_vector_type(4)))  unsigned int u32x4;
typedef __attribute__((ext_vector_type(4)))  int      i32x4;
typedef __attribute__((ext_vector_type(8)))  int      i32x8;

#define PI_D 3.14159265358979323846

// ---- small tables -----------------------------------------------------------
__device__ const int S2tab[8] = {0, 1, 10, 35, 84, 165, 286, 455}; // cum sum (2l+1)^2
// 8-point Gauss-Legendre nodes (ascending) and weights
__device__ const double GLX[8] = {
    -0.9602898564975363, -0.7966664774136267, -0.5255324099163290, -0.1834346424956498,
     0.1834346424956498,  0.5255324099163290,  0.7966664774136267,  0.9602898564975363};
__device__ const double GLW[8] = {
     0.1012285362903763, 0.2223810344533745, 0.3137066458778873, 0.3626837833783620,
     0.3626837833783620, 0.3137066458778873, 0.2223810344533745, 0.1012285362903763};

// ---- workspace layout (bytes) ----------------------------------------------
#define OFF_ROT   ((size_t)0)            // 29120 doubles (EA/EB/EC/PA/PB/PC)
#define OFF_DIN   ((size_t)262144)       // 2048*10 f32
#define OFF_C8W   ((size_t)360448)       // 8*455 f32
#define OFF_BT3   ((size_t)524288)       // 512*2048 f16   (D_OUTW^T, padded rows zeroed)
#define OFF_SIG   ((size_t)4194304)      // 8192*2048 f16
#define OFF_F2F16 ((size_t)37748736)     // 8192*512 f16
#define OFF_TT    ((size_t)46137344)     // 512*8192 f16
#define OFF_PART  ((size_t)54525952)     // 8*512*512 f32  (ends ~62.9 MB)
// rot sub-offsets (in doubles)
#define ROT_EA 0
#define ROT_EB 7280
#define ROT_EC 10920
#define ROT_PA 18200
#define ROT_PB 21840
#define ROT_PC 25480

// ============================================================================
// fp64 helpers: SO(3) real-basis generators + matrix exponential
// ============================================================================
__device__ void dmm(int d, const double* A, const double* B, double* C) {
  for (int i = 0; i < d; ++i)
    for (int j = 0; j < d; ++j) {
      double s = 0.0;
      for (int k = 0; k < d; ++k) s += A[i * d + k] * B[k * d + j];
      C[i * d + j] = s;
    }
}

__device__ void dexpm(int d, double* A, double* E) {  // E = expm(A); A clobbered
  int dd = d * d;
  double nrm = 0.0;
  for (int i = 0; i < d; ++i) {
    double rs = 0.0;
    for (int j = 0; j < d; ++j) rs += fabs(A[i * d + j]);
    nrm = fmax(nrm, rs);
  }
  int s = 0;
  while (nrm > 0.25 && s < 48) { nrm *= 0.5; s++; }
  double sc = ldexp(1.0, -s);
  for (int n = 0; n < dd; ++n) A[n] *= sc;
  double T[169], T2[169];
  for (int n = 0; n < dd; ++n) { E[n] = 0.0; T[n] = 0.0; }
  for (int i = 0; i < d; ++i) { E[i * d + i] = 1.0; T[i * d + i] = 1.0; }
  for (int k = 1; k <= 13; ++k) {
    dmm(d, T, A, T2);
    double inv = 1.0 / (double)k;
    for (int n = 0; n < dd; ++n) { T[n] = T2[n] * inv; E[n] += T[n]; }
  }
  for (int it = 0; it < s; ++it) {
    dmm(d, E, E, T2);
    for (int n = 0; n < dd; ++n) E[n] = T2[n];
  }
}

// Real-basis SO(3) generator: which==0 -> Jx, which==1 -> Jy (e3nn convention)
__device__ void so3_gen(int l, int which, double* G) {
  int d = 2 * l + 1;
  const double RS2 = 0.70710678118654752440;
  double Qr[169], Qi[169];
  for (int n = 0; n < d * d; ++n) { Qr[n] = 0.0; Qi[n] = 0.0; }
  for (int m = -l; m < 0; ++m) {
    Qr[(l + m) * d + (l - m)] = RS2;
    Qi[(l + m) * d + (l + m)] = -RS2;
  }
  Qr[l * d + l] = 1.0;
  for (int m = 1; m <= l; ++m) {
    double sgn = (m & 1) ? -RS2 : RS2;
    Qr[(l + m) * d + (l + m)] = sgn;
    Qi[(l + m) * d + (l - m)] = sgn;
  }
  // global phase (-i)^l
  int ph = l & 3;
  double pr = (ph == 0) ? 1.0 : (ph == 2) ? -1.0 : 0.0;
  double pi = (ph == 1) ? -1.0 : (ph == 3) ? 1.0 : 0.0;
  for (int n = 0; n < d * d; ++n) {
    double qr = Qr[n], qi = Qi[n];
    Qr[n] = qr * pr - qi * pi;
    Qi[n] = qr * pi + qi * pr;
  }
  double Tr[169], Ti[169];
  if (which == 0) {
    for (int a = 0; a < d; ++a)
      for (int b = 0; b < d; ++b) {
        double sr = 0.0, si = 0.0;
        if (a + 1 < d) {
          double s = 0.5 * sqrt((double)(l * (l + 1) - (a - l) * (a - l + 1)));
          sr += s * Qr[(a + 1) * d + b];
          si += s * Qi[(a + 1) * d + b];
        }
        if (a - 1 >= 0) {
          double s = -0.5 * sqrt((double)(l * (l + 1) - (a - 1 - l) * (a - l)));
          sr += s * Qr[(a - 1) * d + b];
          si += s * Qi[(a - 1) * d + b];
        }
        Tr[a * d + b] = sr;
        Ti[a * d + b] = si;
      }
  } else {
    for (int a = 0; a < d; ++a) {
      double m = (double)(a - l);
      for (int b = 0; b < d; ++b) {
        Tr[a * d + b] = -m * Qi[a * d + b];
        Ti[a * d + b] =  m * Qr[a * d + b];
      }
    }
  }
  for (int a = 0; a < d; ++a)
    for (int b = 0; b < d; ++b) {
      double s = 0.0;
      for (int k = 0; k < d; ++k)
        s += Qr[k * d + a] * Tr[k * d + b] + Qi[k * d + a] * Ti[k * d + b];
      G[a * d + b] = s;
    }
}

// escnn C8 element -> Euler angles (a, b, g)
__device__ void c8_angles(int p, double* ao, double* bo, double* go) {
  double th = 2.0 * PI_D * (double)p / 8.0;
  double st = sin(th), ct = cos(th);
  double xb = fmin(1.0, fmax(-1.0, ct));
  double b = acos(xb);
  double a = atan2(st, 0.0);
  double ca = cos(a), sa = sin(a), cb = cos(b), sb = sin(b);
  double P[3][3] = {{ca, sa * sb, sa * cb}, {0.0, cb, -sb}, {-sa, ca * sb, ca * cb}};
  double M[3][3] = {{ct, st, 0.0}, {-st, ct, 0.0}, {0.0, 0.0, 1.0}};
  double R00 = P[0][0] * M[0][0] + P[1][0] * M[1][0] + P[2][0] * M[2][0];
  double R02 = P[0][0] * M[0][2] + P[1][0] * M[1][2] + P[2][0] * M[2][2];
  *ao = a;
  *bo = b;
  *go = atan2(R02, R00);
}

// ============================================================================
// Setup kernels
// ============================================================================
__global__ void k_build_rot(double* rot) {
  int t = blockIdx.x * blockDim.x + threadIdx.x;
  if (t >= 448) return;
  int l = t >> 6, r = t & 63;
  int d = 2 * l + 1, dd = d * d;
  int which;
  double ang;
  double* dst;
  if (r < 16) {
    which = 1; ang = 2.0 * PI_D * (double)r / 16.0;
    dst = rot + ROT_EA + S2tab[l] * 16 + r * dd;
  } else if (r < 24) {
    which = 0; ang = acos(GLX[r - 16]);
    dst = rot + ROT_EB + S2tab[l] * 8 + (r - 16) * dd;
  } else if (r < 40) {
    which = 1; ang = 2.0 * PI_D * (double)(r - 24) / 16.0;
    dst = rot + ROT_EC + S2tab[l] * 16 + (r - 24) * dd;
  } else {
    int p = (r - 40) & 7;
    double a, b, g;
    c8_angles(p, &a, &b, &g);
    if (r < 48)      { which = 1; ang = a; dst = rot + ROT_PA + S2tab[l] * 8 + p * dd; }
    else if (r < 56) { which = 0; ang = b; dst = rot + ROT_PB + S2tab[l] * 8 + p * dd; }
    else             { which = 1; ang = g; dst = rot + ROT_PC + S2tab[l] * 8 + p * dd; }
  }
  double G[169], E[169];
  so3_gen(l, which, G);
  for (int n = 0; n < dd; ++n) G[n] *= ang;
  dexpm(d, G, E);
  for (int n = 0; n < dd; ++n) dst[n] = E[n];
}

__global__ void k_build_c8w(const double* rot, float* C8W) {
  int t = threadIdx.x;
  if (t >= 56) return;
  int l = t >> 3, p = t & 7;
  int d = 2 * l + 1, dd = d * d;
  const double* PA = rot + ROT_PA + S2tab[l] * 8 + p * dd;
  const double* PB = rot + ROT_PB + S2tab[l] * 8 + p * dd;
  const double* PC = rot + ROT_PC + S2tab[l] * 8 + p * dd;
  double T1[169], D[169];
  dmm(d, PA, PB, T1);
  dmm(d, T1, PC, D);
  double sq = sqrt((double)(2 * l + 1));
  int off = S2tab[l];
  for (int m = 0; m < d; ++m)
    for (int q = 0; q < d; ++q)
      C8W[p * 455 + off + m * d + q] = (float)(sq * D[m * d + q]);
}

__global__ void k_build_grids(const double* rot, float* DIN, f16* BT3) {
  int t = blockIdx.x * blockDim.x + threadIdx.x;
  if (t >= 2048 * 7) return;
  int g = t & 2047, l = t >> 11;
  int d = 2 * l + 1, dd = d * d;
  int ic = g & 15, ib = (g >> 4) & 7, ia = g >> 7;
  const double* EA = rot + ROT_EA + S2tab[l] * 16 + ia * dd;
  const double* EB = rot + ROT_EB + S2tab[l] * 8 + ib * dd;
  const double* EC = rot + ROT_EC + S2tab[l] * 16 + ic * dd;
  double T1[169], D[169];
  dmm(d, EA, EB, T1);
  dmm(d, T1, EC, D);
  double sq = sqrt((double)(2 * l + 1));
  if (l <= 1) {
    int off10 = (l == 0) ? 0 : 1;
    for (int m = 0; m < d; ++m)
      for (int q = 0; q < d; ++q)
        DIN[g * 10 + off10 + m * d + q] = (float)(sq * D[m * d + q]);
  }
  double wq = GLW[ib] / 512.0;  // (w_b/2) / (16*16): quadrature folded in
  int off = S2tab[l];
  for (int m = 0; m < d; ++m)
    for (int q = 0; q < d; ++q)
      BT3[(size_t)(off + m * d + q) * 2048 + g] = (f16)(sq * wq * D[m * d + q]);
}

__global__ void k_zero_pad(f16* BT3) {
  int i = blockIdx.x * blockDim.x + threadIdx.x;
  if (i < 57 * 2048) BT3[455 * 2048 + i] = (f16)0.0f;
}

// Fused proj_2 x C8W weights:  TT[n=g*8+i][kk=f*512+j] f16
__global__ void k_build_T(const float* w2_0, const float* w2_1, const float* w2_2,
                          const float* w2_3, const float* w2_4, const float* w2_5,
                          const float* w2_6, const float* C8W, f16* TT) {
  size_t idx = (size_t)blockIdx.x * blockDim.x + threadIdx.x;
  if (idx >= (size_t)512 * 8192) return;
  int n = (int)(idx >> 13);
  int kk = (int)(idx & 8191);
  int g = n >> 3, i = n & 7;
  int f = kk >> 9, j = kk & 511;
  float val = 0.0f;
  if (j < 455) {
    int l = 0;
    while (j >= S2tab[l + 1]) l++;
    int off = S2tab[l], d = 2 * l + 1;
    int u = (j - off) / d, m = (j - off) % d;
    const float* w2;
    switch (l) {
      case 0: w2 = w2_0; break; case 1: w2 = w2_1; break; case 2: w2 = w2_2; break;
      case 3: w2 = w2_3; break; case 4: w2 = w2_4; break; case 5: w2 = w2_5; break;
      default: w2 = w2_6; break;
    }
    float s = 0.0f;
    for (int w = 0; w < d; ++w)
      s += w2[((f * 64 + g) * d + u) * d + w] * C8W[i * 455 + off + w * d + m];
    val = s * rsqrtf(16.0f * (float)d);
  }
  TT[idx] = (f16)val;
}

// ============================================================================
// Stage 1+2: proj_1 + forward grid transform + relu*sqrt2 -> sig f16
// ============================================================================
__global__ void k_sig(const float* __restrict__ traj, const float* __restrict__ w1l0,
                      const float* __restrict__ w1l1, const float* __restrict__ DIN,
                      f16* __restrict__ sig) {
  int bt = blockIdx.x;  // 0..511
  __shared__ float sfeat[160];
  int t = threadIdx.x;
  if (t < 160) {
    int c = t / 10, comp = t % 10;
    float v;
    if (comp == 0) {
      v = traj[bt * 10 + 9] * w1l0[c];
    } else {
      int w = (comp - 1) / 3, m = (comp - 1) % 3;
      float s = 0.0f;
      for (int u = 0; u < 3; ++u) s += traj[bt * 10 + u * 3 + m] * w1l1[(c * 3 + u) * 3 + w];
      v = s * 0.57735026918962576f;  // 3^-1/2
    }
    sfeat[t] = v;
  }
  __syncthreads();
  for (int g = t; g < 2048; g += 256) {
    float din[10];
#pragma unroll
    for (int i = 0; i < 10; ++i) din[i] = DIN[g * 10 + i];
    for (int c = 0; c < 16; ++c) {
      float s = 0.0f;
#pragma unroll
      for (int i = 0; i < 10; ++i) s += sfeat[c * 10 + i] * din[i];
      s = fmaxf(s, 0.0f) * 1.41421356237f;
      sig[(size_t)(bt * 16 + c) * 2048 + g] = (f16)s;
    }
  }
}

// ============================================================================
// TDM 2D tile load: Global -> LDS (pad 16B after every 128B row -> 144B pitch)
// ============================================================================
__device__ __forceinline__ void tdm_load_2d(const void* gptr, unsigned ldsAddr,
                                            unsigned tileCols, unsigned tileRows,
                                            unsigned tenDim0, unsigned tenDim1,
                                            unsigned strideElems) {
  unsigned long long ga = (unsigned long long)gptr;
  u32x4 g0;
  g0[0] = 1u;                                   // count=1, user mode, no gather
  g0[1] = ldsAddr;                              // LDS byte offset
  g0[2] = (unsigned)(ga & 0xFFFFFFFFull);
  g0[3] = (unsigned)((ga >> 32) & 0x01FFFFFFull) | 0x80000000u;  // type=2 ("image")
  i32x8 g1;
  // data_size=2B (code 1) | pad_enable | pad_interval=32 DW (code 4) | pad_amount=4 DW (code 3)
  g1[0] = (int)((1u << 16) | (1u << 20) | (4u << 22) | (3u << 25));
  g1[1] = (int)((tenDim0 & 0xFFFFu) << 16);
  g1[2] = (int)((tenDim0 >> 16) | ((tenDim1 & 0xFFFFu) << 16));
  g1[3] = (int)((tenDim1 >> 16) | ((tileCols & 0xFFFFu) << 16));
  g1[4] = (int)(tileRows & 0xFFFFu);
  g1[5] = (int)strideElems;
  g1[6] = 0;
  g1[7] = 0;
  i32x4 g2 = {0, 0, 0, 0};
  i32x4 g3 = {0, 0, 0, 0};
#if defined(__clang_major__) && __clang_major__ >= 23
  i32x8 g4 = {0, 0, 0, 0, 0, 0, 0, 0};
  __builtin_amdgcn_tensor_load_to_lds(g0, g1, g2, g3, g4, 0);
#else
  __builtin_amdgcn_tensor_load_to_lds(g0, g1, g2, g3, 0);
#endif
}

// ============================================================================
// NT-GEMM: C[M,N] = A[M,K](f16,row-major) * BT[N,K](f16,row-major)^T
// 128x128 tile / WG (4 waves, 64x64 each), K-tile 64, TDM double-buffered LDS,
// v_wmma_f32_16x16x32_f16. blockIdx.z = split-K partial (f32 out);
// writeHalf!=0 -> single-split, f16 epilogue directly to Ch.
// ============================================================================
#define RS 144   // LDS row pitch (bytes): 128B data + 16B TDM pad (conflict-free)
#define TILEB (128 * RS)

__global__ __launch_bounds__(128) void k_gemm_nt(const f16* __restrict__ A,
                                                 const f16* __restrict__ BT,
                                                 float* __restrict__ C,
                                                 f16* __restrict__ Ch, int M, int N,
                                                 int K, int kPerSplit, int writeHalf) {
  __shared__ __align__(1024) char smem[4 * TILEB];  // buf0{A,B}, buf1{A,B}
  unsigned ldsBase = (unsigned)(unsigned long long)(void*)smem;

  int tid = threadIdx.x;
  int wave = tid >> 5, lane = tid & 31;
  int lh = lane >> 4, lr = lane & 15;
  int wm = (wave >> 1) << 6, wn = (wave & 1) << 6;
  long long m0 = (long long)blockIdx.y * 128;
  long long n0 = (long long)blockIdx.x * 128;
  long long k0 = (long long)blockIdx.z * kPerSplit;
  int niter = kPerSplit >> 6;

  v8f acc[4][4] = {};

  // pre-issue buffer 0 (wave0 only; TDM issues once per wave, EXEC-independent)
  if (wave == 0) {
    tdm_load_2d(A + m0 * K + k0, ldsBase, 64u, 128u, (unsigned)K, (unsigned)M,
                (unsigned)K);
    tdm_load_2d(BT + n0 * K + k0, ldsBase + TILEB, 64u, 128u, (unsigned)K, (unsigned)N,
                (unsigned)K);
  }

  for (int i = 0; i < niter; ++i) {
    int cur = i & 1;
    if (wave == 0) {
      if (i + 1 < niter) {
        long long kn = k0 + (long long)(i + 1) * 64;
        unsigned nb = ldsBase + (unsigned)((i + 1) & 1) * (2u * TILEB);
        tdm_load_2d(A + m0 * K + kn, nb, 64u, 128u, (unsigned)K, (unsigned)M,
                    (unsigned)K);
        tdm_load_2d(BT + n0 * K + kn, nb + TILEB, 64u, 128u, (unsigned)K, (unsigned)N,
                    (unsigned)K);
        // TDM ops retire in order per wave: <=2 outstanding => current buf complete
        __builtin_amdgcn_s_wait_tensorcnt(2);
      } else {
        __builtin_amdgcn_s_wait_tensorcnt(0);
      }
    }
    __syncthreads();
    const char* bA = smem + cur * (2 * TILEB);
    const char* bB = bA + TILEB;
#pragma unroll
    for (int kc = 0; kc < 2; ++kc) {
      v16h af[4], bf[4];
#pragma unroll
      for (int mt = 0; mt < 4; ++mt) {
        // A fragment: lane (lh,lr) -> row wm+mt*16+lr, K = {kc*32+lh*8..+7} u {+16..}
        const char* p = bA + (wm + mt * 16 + lr) * RS + (kc * 32 + lh * 8) * 2;
        v8h lo = *(const v8h*)p;
        v8h hi = *(const v8h*)(p + 32);
        af[mt] = __builtin_shufflevector(lo, hi, 0, 1, 2, 3, 4, 5, 6, 7, 8, 9, 10, 11,
                                         12, 13, 14, 15);
      }
#pragma unroll
      for (int nt = 0; nt < 4; ++nt) {
        // B fragment: lane (lh,lr) -> column n=wn+nt*16+lr, K = {kc*32+lh*8..+7} u {+16..}
        const char* q = bB + (wn + nt * 16 + lr) * RS + (kc * 32 + lh * 8) * 2;
        v8h lo = *(const v8h*)q;
        v8h hi = *(const v8h*)(q + 32);
        bf[nt] = __builtin_shufflevector(lo, hi, 0, 1, 2, 3, 4, 5, 6, 7, 8, 9, 10, 11,
                                         12, 13, 14, 15);
      }
#pragma unroll
      for (int mt = 0; mt < 4; ++mt)
#pragma unroll
        for (int nt = 0; nt < 4; ++nt)
          acc[mt][nt] = __builtin_amdgcn_wmma_f32_16x16x32_f16(
              false, af[mt], false, bf[nt], (short)0, acc[mt][nt], false, false);
    }
    __syncthreads();  // guards next iteration's TDM write into buf (i+1)&1's peer
  }

  if (writeHalf) {
#pragma unroll
    for (int mt = 0; mt < 4; ++mt)
#pragma unroll
      for (int nt = 0; nt < 4; ++nt) {
        long long rb = m0 + wm + mt * 16 + lh * 8;
        long long cb = n0 + wn + nt * 16 + lr;
#pragma unroll
        for (int i = 0; i < 8; ++i) Ch[(rb + i) * N + cb] = (f16)acc[mt][nt][i];
      }
  } else {
    float* Cp = C + (long long)blockIdx.z * M * N;
#pragma unroll
    for (int mt = 0; mt < 4; ++mt)
#pragma unroll
      for (int nt = 0; nt < 4; ++nt) {
        long long rb = m0 + wm + mt * 16 + lh * 8;
        long long cb = n0 + wn + nt * 16 + lr;
#pragma unroll
        for (int i = 0; i < 8; ++i) Cp[(rb + i) * N + cb] = acc[mt][nt][i];
      }
  }
}

// ============================================================================
// Epilogues
// ============================================================================
__global__ void k_reduce(const float* __restrict__ part, float* __restrict__ out) {
  int i = blockIdx.x * blockDim.x + threadIdx.x;
  if (i < 262144) {
    float s = 0.0f;
#pragma unroll
    for (int z = 0; z < 8; ++z) s += part[z * 262144 + i];
    out[i] = s;
  }
}

// x_c8: memory-bound 65536x455 @ C8W^T (8 outputs/row), coalesced f32 + prefetch
__global__ void k_xc8(const float* __restrict__ x, const float* __restrict__ C8W,
                      float* __restrict__ out) {
  __shared__ float sc[455 * 8];  // [j][i]
  int t = threadIdx.x;
  for (int idx = t; idx < 455 * 8; idx += 256) {
    int j = idx >> 3, i = idx & 7;
    sc[idx] = C8W[i * 455 + j];
  }
  __syncthreads();
  int wave = t >> 5, lane = t & 31;
  long long stride = (long long)gridDim.x * 8;
  for (long long row = (long long)blockIdx.x * 8 + wave; row < 65536; row += stride) {
    long long nrow = row + stride;
    if (nrow < 65536) __builtin_prefetch(x + nrow * 455 + lane, 0, 1);
    float acc[8] = {0, 0, 0, 0, 0, 0, 0, 0};
    for (int j = lane; j < 455; j += 32) {
      float xv = x[row * 455 + j];
      const float* cj = &sc[j * 8];
#pragma unroll
      for (int i = 0; i < 8; ++i) acc[i] += xv * cj[i];
    }
#pragma unroll
    for (int i = 0; i < 8; ++i)
      for (int off = 16; off > 0; off >>= 1) acc[i] += __shfl_xor(acc[i], off, 32);
    if (lane == 0) {
#pragma unroll
      for (int i = 0; i < 8; ++i) out[row * 8 + i] = acc[i];
    }
  }
}

// ============================================================================
// Launcher
// ============================================================================
extern "C" void kernel_launch(void* const* d_in, const int* in_sizes, int n_in,
                              void* d_out, int out_size, void* d_ws, size_t ws_size,
                              hipStream_t stream) {
  const float* x    = (const float*)d_in[0];
  const float* traj = (const float*)d_in[1];
  const float* w1l0 = (const float*)d_in[2];
  const float* w1l1 = (const float*)d_in[3];
  const float* w2[7];
  for (int i = 0; i < 7; ++i) w2[i] = (const float*)d_in[4 + i];

  char* ws = (char*)d_ws;
  double* rot  = (double*)(ws + OFF_ROT);
  float* DIN   = (float*)(ws + OFF_DIN);
  float* C8W   = (float*)(ws + OFF_C8W);
  f16* BT3     = (f16*)(ws + OFF_BT3);
  f16* SIG     = (f16*)(ws + OFF_SIG);
  f16* F2H     = (f16*)(ws + OFF_F2F16);
  f16* TT      = (f16*)(ws + OFF_TT);
  float* PART  = (float*)(ws + OFF_PART);
  float* out_x = (float*)d_out;       // (16,32,128,8) = 524288
  float* out_t = out_x + 524288;      // (16,32,64,8)  = 262144

  // constants (device-side rebuild each call; deterministic)
  k_build_rot<<<2, 256, 0, stream>>>(rot);
  k_build_c8w<<<1, 64, 0, stream>>>(rot, C8W);
  k_build_grids<<<(2048 * 7 + 255) / 256, 256, 0, stream>>>(rot, DIN, BT3);
  k_zero_pad<<<(57 * 2048 + 255) / 256, 256, 0, stream>>>(BT3);
  k_build_T<<<(512 * 8192) / 256, 256, 0, stream>>>(w2[0], w2[1], w2[2], w2[3], w2[4],
                                                    w2[5], w2[6], C8W, TT);
  // trajectory path
  k_sig<<<512, 256, 0, stream>>>(traj, w1l0, w1l1, DIN, SIG);
  k_gemm_nt<<<dim3(4, 64, 1), 128, 0, stream>>>(SIG, BT3, (float*)nullptr, F2H, 8192,
                                                512, 2048, 2048, 1);
  k_gemm_nt<<<dim3(4, 4, 8), 128, 0, stream>>>(F2H, TT, PART, (f16*)nullptr, 512, 512,
                                               8192, 1024, 0);
  k_reduce<<<262144 / 256, 256, 0, stream>>>(PART, out_t);
  // x path (memory-bound)
  k_xc8<<<1024, 256, 0, stream>>>(x, C8W, out_x);
}
